// PolicyHead_52106543235080
// MI455X (gfx1250) — matrix operand order repaired and checked by reading, not compiled, not run on hardware
//
#include <hip/hip_runtime.h>
#include <math.h>

// ---------------------------------------------------------------------------
// PolicyHead for MI455X (gfx1250, wave32, WMMA).
//  q = backbone(16384x4096) @ Wq(4096x256) + bq    -> bf16 hi/lo split WMMA
//  k = LayerNorm(key_table[aidx])                  -> precomputed 81x256 table
//  attn = q.k/16 (+bias on action0) -> log_softmax -> out (16384x40) f32
// Roofline: 34.4 GFLOP GEMM over 256MB activations -> memory-bound at
// 23.3 TB/s (~11us) provided matrix ops run on the bf16 WMMA path.
// ---------------------------------------------------------------------------

typedef __bf16 bf16_t;
typedef __attribute__((ext_vector_type(16))) __bf16 v16bf;
typedef __attribute__((ext_vector_type(8)))  __bf16 v8bf;
typedef __attribute__((ext_vector_type(4)))  __bf16 v4bf;
typedef __attribute__((ext_vector_type(8)))  float  v8f;

#define D_IN   4096
#define D_Q    256
#define NUM_A  40
#define NACT   81          // key table rows
#define M_BLK  64          // rows of q per block
#define KCH    64          // staged K per chunk = two K=32 WMMA steps
#define AST    72          // A LDS row stride (bf16), 144B: 16B aligned, bank-spread
#define BST    72          // B^T LDS row stride (bf16)
#define QST    264         // q LDS row stride (floats)
#define KLST   260         // K_ln LDS row stride (floats)

// ---- CDNA5 async global->LDS path (probe-guarded) -------------------------
// Diagnostic from previous round: param type is 'vector_size(16) int *'.
#if __has_builtin(__builtin_amdgcn_global_load_async_to_lds_b128) && \
    __has_builtin(__builtin_amdgcn_s_wait_asynccnt)
#define HAVE_ASYNC_LDS 1
typedef int v4i_ __attribute__((vector_size(16)));
typedef __attribute__((address_space(1))) v4i_ as1_v4i;
typedef __attribute__((address_space(3))) v4i_ as3_v4i;
__device__ __forceinline__ void cp16_async(void* lds, const void* g) {
  __builtin_amdgcn_global_load_async_to_lds_b128((as1_v4i*)g, (as3_v4i*)lds, 0, 0);
}
#else
#define HAVE_ASYNC_LDS 0
#endif

// ---------------------------------------------------------------------------
// Prep 1: Wq (K x N, f32) -> WqT_hi / WqT_lo (N x K, bf16) via LDS transpose.
// ---------------------------------------------------------------------------
__global__ __launch_bounds__(256) void prep_wq(const float* __restrict__ Wq,
                                               bf16_t* __restrict__ wqt_hi,
                                               bf16_t* __restrict__ wqt_lo) {
  __shared__ float tile[32][33];
  const int t  = threadIdx.x;
  const int k0 = blockIdx.x * 32;
  const int n0 = blockIdx.y * 32;
  {
    const int rr = t >> 3;
    const int cc = (t & 7) * 4;
    float4 v = *(const float4*)(Wq + (size_t)(k0 + rr) * D_Q + n0 + cc);
    tile[rr][cc + 0] = v.x; tile[rr][cc + 1] = v.y;
    tile[rr][cc + 2] = v.z; tile[rr][cc + 3] = v.w;
  }
  __syncthreads();
  {
    const int nn = t >> 3;
    const int kk = (t & 7) * 4;
    v4bf h, l;
    for (int i = 0; i < 4; ++i) {
      float x = tile[kk + i][nn];
      __bf16 hi = (__bf16)x;
      h[i] = hi;
      l[i] = (__bf16)(x - (float)hi);
    }
    *(v4bf*)(wqt_hi + (size_t)(n0 + nn) * D_IN + k0 + kk) = h;
    *(v4bf*)(wqt_lo + (size_t)(n0 + nn) * D_IN + k0 + kk) = l;
  }
}

// ---------------------------------------------------------------------------
// Prep 2: K_ln[row] = LayerNorm(key_table[row]) * gamma + beta   (81 x 256)
// ---------------------------------------------------------------------------
__global__ __launch_bounds__(256) void prep_keys(const float* __restrict__ kt,
                                                 const float* __restrict__ gamma,
                                                 const float* __restrict__ beta,
                                                 float* __restrict__ kln) {
  const int row = blockIdx.x;
  const int d   = threadIdx.x;
  float x = kt[row * D_Q + d];
  float s = x, ss = x * x;
  for (int m = 16; m; m >>= 1) {
    s  += __shfl_xor(s, m, 32);
    ss += __shfl_xor(ss, m, 32);
  }
  __shared__ float ps[8], pss[8];
  if ((d & 31) == 0) { ps[d >> 5] = s; pss[d >> 5] = ss; }
  __syncthreads();
  float S = 0.f, SS = 0.f;
  for (int i = 0; i < 8; ++i) { S += ps[i]; SS += pss[i]; }
  const float mu  = S * (1.0f / 256.0f);
  const float var = SS * (1.0f / 256.0f) - mu * mu;
  const float inv = rsqrtf(var + 1e-5f);
  kln[row * D_Q + d] = (x - mu) * inv * gamma[d] + beta[d];
}

// ---------------------------------------------------------------------------
// Main fused kernel.  Block = 256 threads = 8 waves laid out 2(M) x 4(N).
// Each wave owns 2 M-tiles x 4 N-tiles (covers the full 64x256 block tile).
// ---------------------------------------------------------------------------
__global__ __launch_bounds__(256) void policy_main(
    const float*  __restrict__ backbone,
    const bf16_t* __restrict__ wqt_hi,
    const bf16_t* __restrict__ wqt_lo,
    const float*  __restrict__ bq,
    const float*  __restrict__ kln_g,
    const int*    __restrict__ va,
    const int*    __restrict__ phase,
    const int*    __restrict__ trick,
    const float*  __restrict__ p_signal,
    float*        __restrict__ out) {
  // GEMM staging region, later reused for the LayerNorm'd key table.
  __shared__ __align__(16) unsigned char smu[92160];        // 90 KB
  __shared__ __align__(16) float q_lds[M_BLK * QST];        // 66 KB

  bf16_t* A_hi = (bf16_t*)smu;                 // 64 x 72 bf16
  bf16_t* A_lo = A_hi + M_BLK * AST;
  bf16_t* B_hi = A_lo + M_BLK * AST;           // 256 x 72 bf16 (B^T: [n][k])
  bf16_t* B_lo = B_hi + D_Q * BST;

  const int tid  = threadIdx.x;
  const int wave = tid >> 5;
  const int lane = tid & 31;
  const int hl   = lane >> 4;                  // half-wave
  const int lm   = lane & 15;
  const int mw2  = wave & 1;                   // M group (2 tiles each)
  const int nq   = wave >> 1;                  // N group (4 tiles each)
  const int row0 = blockIdx.x * M_BLK;

  v8f acc[2][4];
  for (int mi = 0; mi < 2; ++mi)
    for (int j = 0; j < 4; ++j)
      for (int i = 0; i < 8; ++i) acc[mi][j][i] = 0.0f;

  const int sa_r = tid >> 2;                   // staging: 64 rows x 16 floats
  const int sa_k = (tid & 3) * 16;
  const float*  aptr    = backbone + (size_t)(row0 + sa_r) * D_IN + sa_k;
  const bf16_t* bptr_hi = wqt_hi + (size_t)tid * D_IN;   // one W^T row / thread
  const bf16_t* bptr_lo = wqt_lo + (size_t)tid * D_IN;

  for (int k0 = 0; k0 < D_IN; k0 += KCH) {
    __syncthreads();
    // -- B chunk: bf16 copy of L2-resident pre-transposed WqT --------------
#if HAVE_ASYNC_LDS
    {
      bf16_t* dh = B_hi + tid * BST;
      bf16_t* dl = B_lo + tid * BST;
      #pragma unroll
      for (int c = 0; c < 8; ++c) {            // 8 x 16B per operand
        cp16_async(dh + c * 8, bptr_hi + k0 + c * 8);
        cp16_async(dl + c * 8, bptr_lo + k0 + c * 8);
      }
    }
#else
    {
      const v8bf* sh = (const v8bf*)(bptr_hi + k0);
      const v8bf* sl = (const v8bf*)(bptr_lo + k0);
      v8bf* dh = (v8bf*)(B_hi + tid * BST);
      v8bf* dl = (v8bf*)(B_lo + tid * BST);
      #pragma unroll
      for (int c = 0; c < 8; ++c) { dh[c] = sh[c]; dl[c] = sl[c]; }
    }
#endif
    // -- A chunk: fp32 -> (hi,lo) bf16 split, shared by all 8 waves --------
    {
      const float* src = aptr + k0;
      float xs[16];
      #pragma unroll
      for (int c = 0; c < 4; ++c) {
        float4 v = *(const float4*)(src + c * 4);
        xs[c * 4 + 0] = v.x; xs[c * 4 + 1] = v.y;
        xs[c * 4 + 2] = v.z; xs[c * 4 + 3] = v.w;
      }
      v8bf h0, h1, l0, l1;
      #pragma unroll
      for (int i = 0; i < 8; ++i) {
        __bf16 hi = (__bf16)xs[i];
        h0[i] = hi;
        l0[i] = (__bf16)(xs[i] - (float)hi);
        __bf16 hj = (__bf16)xs[8 + i];
        h1[i] = hj;
        l1[i] = (__bf16)(xs[8 + i] - (float)hj);
      }
      *(v8bf*)(A_hi + sa_r * AST + sa_k)     = h0;
      *(v8bf*)(A_hi + sa_r * AST + sa_k + 8) = h1;
      *(v8bf*)(A_lo + sa_r * AST + sa_k)     = l0;
      *(v8bf*)(A_lo + sa_r * AST + sa_k + 8) = l1;
      if (k0 + KCH < D_IN) __builtin_prefetch(src + KCH, 0, 3);
    }
#if HAVE_ASYNC_LDS
    __builtin_amdgcn_s_wait_asynccnt(0);
#endif
    __syncthreads();
    // -- compute: two K=32 WMMA steps over the staged 64-deep chunk --------
    #pragma unroll
    for (int s = 0; s < 2; ++s) {
      union { v16bf v; v8bf h[2]; } ah[2], al[2];
      #pragma unroll
      for (int mi = 0; mi < 2; ++mi) {
        // A fragment (M=lm, K=(hl?16:0)+e contiguous)
        const int ar = mw2 * 32 + mi * 16 + lm;
        const bf16_t* ph = A_hi + ar * AST + s * 32;
        const bf16_t* pl = A_lo + ar * AST + s * 32;
        ah[mi].h[0] = *(const v8bf*)(ph + hl * 16);
        ah[mi].h[1] = *(const v8bf*)(ph + hl * 16 + 8);
        al[mi].h[0] = *(const v8bf*)(pl + hl * 16);
        al[mi].h[1] = *(const v8bf*)(pl + hl * 16 + 8);
      }
      #pragma unroll
      for (int j = 0; j < 4; ++j) {
        // B fragment (N=lm, K=(hl?16:0)+e contiguous in B^T row)
        const int n = nq * 64 + j * 16 + lm;
        const bf16_t* pbh = B_hi + n * BST + s * 32 + hl * 16;
        const bf16_t* pbl = B_lo + n * BST + s * 32 + hl * 16;
        union { v16bf v; v8bf h[2]; } bh, bl;
        bh.h[0] = *(const v8bf*)(pbh); bh.h[1] = *(const v8bf*)(pbh + 8);
        bl.h[0] = *(const v8bf*)(pbl); bl.h[1] = *(const v8bf*)(pbl + 8);
        #pragma unroll
        for (int mi = 0; mi < 2; ++mi) {
          acc[mi][j] = __builtin_amdgcn_wmma_f32_16x16x32_bf16(
              false, ah[mi].v, false, bh.v, (short)0, acc[mi][j], false, false);
          acc[mi][j] = __builtin_amdgcn_wmma_f32_16x16x32_bf16(
              false, ah[mi].v, false, bl.v, (short)0, acc[mi][j], false, false);
          acc[mi][j] = __builtin_amdgcn_wmma_f32_16x16x32_bf16(
              false, al[mi].v, false, bh.v, (short)0, acc[mi][j], false, false);
        }
      }
    }
  }
  __syncthreads();   // all waves done with A/B LDS before repurposing it

  // -- spill q tile (+bq) to LDS; C layout: M = vgpr + hl*8, N = lm --------
  for (int mi = 0; mi < 2; ++mi)
    for (int j = 0; j < 4; ++j) {
      const int col = nq * 64 + j * 16 + lm;
      const float b = bq[col];
      for (int r = 0; r < 8; ++r)
        q_lds[(mw2 * 32 + mi * 16 + hl * 8 + r) * QST + col] = acc[mi][j][r] + b;
    }
  // -- stage LayerNorm'd key table into (padded) LDS -----------------------
  float* klds = (float*)smu;
  for (int e = tid; e < NACT * (D_Q / 4); e += 256) {
    const int r = e >> 6;             // 64 float4 per row
    const int c = (e & 63) << 2;
    *(float4*)(klds + r * KLST + c) = *(const float4*)(kln_g + r * D_Q + c);
  }
  __syncthreads();

  // -- attention + bias + log_softmax: wave = 8 rows, lane = action --------
  const float INFV = __builtin_inff();
  for (int i = 0; i < 8; ++i) {
    const int r  = wave * 8 + i;
    const int gr = row0 + r;
    const int a1 = lane;
    const bool has2 = lane < (NUM_A - 32);     // lanes 0..7 own action 32+lane
    const int a2 = lane + 32;

    const int rank1 = va[(size_t)gr * (2 * NUM_A) + 2 * a1];
    const int suit1 = va[(size_t)gr * (2 * NUM_A) + 2 * a1 + 1];
    int rank2 = -1, suit2 = 0;
    if (has2) {
      rank2 = va[(size_t)gr * (2 * NUM_A) + 2 * a2];
      suit2 = va[(size_t)gr * (2 * NUM_A) + 2 * a2 + 1];
    }
    const int ph = phase[gr];

    float cnt = (rank1 != -1 ? 1.f : 0.f) + (rank2 != -1 ? 1.f : 0.f);
    for (int m = 16; m; m >>= 1) cnt += __shfl_xor(cnt, m, 32);

    const float* qr = q_lds + r * QST;
    float attn1 = -INFV, attn2 = -INFV;
    if (rank1 != -1) {
      const int idx = (suit1 == 4) ? 80 : (ph * 40 + suit1 * 9 + rank1);
      const float* kr = klds + idx * KLST;
      float s = 0.f;
      for (int d = 0; d < D_Q; d += 4) {
        float4 qv = *(const float4*)(qr + d);
        float4 kv = *(const float4*)(kr + d);
        s += qv.x * kv.x + qv.y * kv.y + qv.z * kv.z + qv.w * kv.w;
      }
      attn1 = s * 0.0625f;
    }
    if (rank2 != -1) {
      const int idx = (suit2 == 4) ? 80 : (ph * 40 + suit2 * 9 + rank2);
      const float* kr = klds + idx * KLST;
      float s = 0.f;
      for (int d = 0; d < D_Q; d += 4) {
        float4 qv = *(const float4*)(qr + d);
        float4 kv = *(const float4*)(kr + d);
        s += qv.x * kv.x + qv.y * kv.y + qv.z * kv.z + qv.w * kv.w;
      }
      attn2 = s * 0.0625f;
    }
    if (a1 == 0 && ph == 1) {                  // no-signal bias on action 0
      const float psv = p_signal[trick[gr]];
      float w = logf(fmaxf((1.f - psv) / psv * (cnt - 1.f), 1e-5f));
      if (cnt == 1.f) w = 0.f;
      attn1 += w;
    }
    float mx = fmaxf(attn1, attn2);
    for (int m = 16; m; m >>= 1) mx = fmaxf(mx, __shfl_xor(mx, m, 32));
    float es = expf(attn1 - mx) + expf(attn2 - mx);   // exp(-inf)=0 if invalid
    for (int m = 16; m; m >>= 1) es += __shfl_xor(es, m, 32);
    const float lse = logf(es);
    out[(size_t)gr * NUM_A + a1] = attn1 - mx - lse;
    if (has2) out[(size_t)gr * NUM_A + a2] = attn2 - mx - lse;
  }
}

// ---------------------------------------------------------------------------
extern "C" void kernel_launch(void* const* d_in, const int* in_sizes, int n_in,
                              void* d_out, int out_size, void* d_ws, size_t ws_size,
                              hipStream_t stream) {
  const float* backbone = (const float*)d_in[0];
  const int*   va       = (const int*)d_in[1];
  const int*   phase    = (const int*)d_in[2];
  const int*   trick    = (const int*)d_in[3];
  const float* Wq       = (const float*)d_in[4];
  const float* bq       = (const float*)d_in[5];
  const float* keytab   = (const float*)d_in[6];
  const float* gamma    = (const float*)d_in[7];
  const float* beta     = (const float*)d_in[8];
  const float* psig     = (const float*)d_in[9];
  float* out = (float*)d_out;

  const int B = in_sizes[0] / D_IN;            // 16384

  // workspace: WqT_hi (2MB) | WqT_lo (2MB) | K_ln (83KB)
  bf16_t* wqt_hi = (bf16_t*)d_ws;
  bf16_t* wqt_lo = wqt_hi + (size_t)D_Q * D_IN;
  float*  kln    = (float*)(wqt_lo + (size_t)D_Q * D_IN);

  prep_wq<<<dim3(D_IN / 32, D_Q / 32), 256, 0, stream>>>(Wq, wqt_hi, wqt_lo);
  prep_keys<<<NACT, 256, 0, stream>>>(keytab, gamma, beta, kln);
  policy_main<<<B / M_BLK, 256, 0, stream>>>(backbone, wqt_hi, wqt_lo, bq, kln,
                                             va, phase, trick, psig, out);
}
